// SparseConvNeXtV1_19473381720397
// MI455X (gfx1250) — matrix operand order repaired and checked by reading, not compiled, not run on hardware
//
#include <hip/hip_runtime.h>
#include <cmath>

// ---------------------------------------------------------------------------
// SparseConvNeXtV1 forward on MI455X (gfx1250).
// MLP GEMMs (≈270 GFLOP total, the dominant cost) run on v_wmma_f32_16x16x32_bf16.
// Wave tile = 16 x (16*NT): NT=4 wherever N%64==0 (all GEMM1s, GEMM2 stages 2-4)
// to amortize A-fragment loads 4x; NT=2 fallback for N=96.
// The K-loop is software double-buffered: loads for step k+32 are issued before
// the WMMAs of step k, so waits are partial (loadcnt<=N) instead of drain-all.
// Elementwise work (dwconv7x7, LayerNorm, masks, transposes) is bandwidth-trivial
// (whole per-stage working set <= 19 MB, fits L2 = 192 MB).
// Wave32 everywhere; LN uses 32-lane butterfly reductions.
// ---------------------------------------------------------------------------

#define DEV __device__ __forceinline__

typedef __attribute__((ext_vector_type(16))) __bf16          v16bf;
typedef __attribute__((ext_vector_type(8)))  float           v8f;
typedef __attribute__((ext_vector_type(8)))  unsigned short  us8;

union Frag { v16bf v; us8 h[2]; };

DEV unsigned short f2bf(float f) {              // round-to-nearest-even fp32->bf16
  unsigned int u = __builtin_bit_cast(unsigned int, f);
  u += 0x7FFFu + ((u >> 16) & 1u);
  return (unsigned short)(u >> 16);
}

DEV float gelu_exact(float x) { return 0.5f * x * (1.0f + erff(x * 0.7071067811865476f)); }

DEV float wave_sum32(float v) {                 // wave32 butterfly reduction
  v += __shfl_xor(v, 16, 32);
  v += __shfl_xor(v, 8, 32);
  v += __shfl_xor(v, 4, 32);
  v += __shfl_xor(v, 2, 32);
  v += __shfl_xor(v, 1, 32);
  return v;
}

// ---------------------------------------------------------------------------
// Active-mask kernels. active in {0,1} as float, one value per spatial point.
// ---------------------------------------------------------------------------
__global__ void active0_kernel(const int* __restrict__ mask, float* __restrict__ act, int Bn) {
  int idx = blockIdx.x * blockDim.x + threadIdx.x;
  int total = Bn * 56 * 56;
  if (idx >= total) return;
  int j = idx % 56, t = idx / 56;
  int i = t % 56, b = t / 56;
  act[idx] = 1.0f - (float)mask[b * 49 + (i >> 3) * 7 + (j >> 3)];
}

__global__ void down_active_kernel(const float* __restrict__ a, float* __restrict__ o,
                                   int Bn, int Ho, int Wo) {
  int idx = blockIdx.x * blockDim.x + threadIdx.x;
  int total = Bn * Ho * Wo;
  if (idx >= total) return;
  int j = idx % Wo, t = idx / Wo;
  int i = t % Ho, b = t / Ho;
  int Wi = Wo * 2;
  const float* s = a + (size_t)b * (Ho * 2) * Wi;
  float m0 = s[(2 * i) * Wi + 2 * j],     m1 = s[(2 * i) * Wi + 2 * j + 1];
  float m2 = s[(2 * i + 1) * Wi + 2 * j], m3 = s[(2 * i + 1) * Wi + 2 * j + 1];
  o[idx] = fmaxf(fmaxf(m0, m1), fmaxf(m2, m3));
}

// ---------------------------------------------------------------------------
// Stem: NCHW x [16,3,224,224] -> conv4x4 stride4 -> NHWC h [16,56,56,96] + bias.
// ---------------------------------------------------------------------------
__global__ void stem_kernel(const float* __restrict__ x, const float* __restrict__ w,
                            const float* __restrict__ bias, float* __restrict__ h, int Bn) {
  const int H = 56, W = 56, C = 96;
  int idx = blockIdx.x * blockDim.x + threadIdx.x;
  int total = Bn * H * W * C;
  if (idx >= total) return;
  int co = idx % C, p = idx / C;
  int j = p % W, t = p / W;
  int i = t % H, b = t / H;
  float s = bias[co];
  for (int ci = 0; ci < 3; ++ci)
    for (int u = 0; u < 4; ++u)
      for (int v = 0; v < 4; ++v) {
        float xv = x[(((size_t)b * 3 + ci) * 224 + (i * 4 + u)) * 224 + (j * 4 + v)];
        s += xv * w[((u * 4 + v) * 3 + ci) * C + co];
      }
  h[idx] = s;
}

// ---------------------------------------------------------------------------
// Per-point channel LayerNorm (biased var, eps 1e-6). One wave32 per point.
// f32 variant multiplies by per-point active; bf16 variant converts for WMMA.
// ---------------------------------------------------------------------------
__global__ void ln_f32_kernel(const float* __restrict__ src, float* __restrict__ dst,
                              const float* __restrict__ g, const float* __restrict__ b,
                              const float* __restrict__ act, int P, int C) {
  int wave = threadIdx.x >> 5, lane = threadIdx.x & 31;
  int p = blockIdx.x * (blockDim.x >> 5) + wave;
  if (p >= P) return;
  const float* row = src + (size_t)p * C;
  float s = 0.f, ss = 0.f;
  for (int c = lane; c < C; c += 32) { float v = row[c]; s += v; ss += v * v; }
  s = wave_sum32(s); ss = wave_sum32(ss);
  float mu = s / (float)C;
  float rs = rsqrtf(ss / (float)C - mu * mu + 1e-6f);
  float a = act[p];
  float* drow = dst + (size_t)p * C;
  for (int c = lane; c < C; c += 32)
    drow[c] = ((row[c] - mu) * rs * g[c] + b[c]) * a;
}

__global__ void ln_bf16_kernel(const float* __restrict__ src, unsigned short* __restrict__ dst,
                               const float* __restrict__ g, const float* __restrict__ b,
                               int P, int C) {
  int wave = threadIdx.x >> 5, lane = threadIdx.x & 31;
  int p = blockIdx.x * (blockDim.x >> 5) + wave;
  if (p >= P) return;
  const float* row = src + (size_t)p * C;
  float s = 0.f, ss = 0.f;
  for (int c = lane; c < C; c += 32) { float v = row[c]; s += v; ss += v * v; }
  s = wave_sum32(s); ss = wave_sum32(ss);
  float mu = s / (float)C;
  float rs = rsqrtf(ss / (float)C - mu * mu + 1e-6f);
  unsigned short* drow = dst + (size_t)p * C;
  for (int c = lane; c < C; c += 32)
    drow[c] = f2bf((row[c] - mu) * rs * g[c] + b[c]);
}

// ---------------------------------------------------------------------------
// 7x7 depthwise conv, SAME padding. h is zero at inactive sites (invariant),
// which realizes the "sparse dwconv" semantics of the reference.
// ---------------------------------------------------------------------------
__global__ void dwconv7_kernel(const float* __restrict__ h, const float* __restrict__ w,
                               const float* __restrict__ bias, float* __restrict__ y,
                               int Bn, int H, int W, int C) {
  size_t idx = (size_t)blockIdx.x * blockDim.x + threadIdx.x;
  size_t total = (size_t)Bn * H * W * C;
  if (idx >= total) return;
  int c = (int)(idx % C); size_t p = idx / C;
  int jx = (int)(p % W); size_t t = p / W;
  int iy = (int)(t % H); int b = (int)(t / H);
  float s = bias[c];
  for (int u = 0; u < 7; ++u) {
    int yy = iy + u - 3;
    if (yy < 0 || yy >= H) continue;
    for (int v = 0; v < 7; ++v) {
      int xx = jx + v - 3;
      if (xx < 0 || xx >= W) continue;
      s += h[(((size_t)b * H + yy) * W + xx) * C + c] * w[(u * 7 + v) * C + c];
    }
  }
  y[idx] = s;
}

// ---------------------------------------------------------------------------
// fp32 [K][N] weight -> bf16 [N][K] (transposed so B fragments are contiguous).
// ---------------------------------------------------------------------------
__global__ void tcvt_kernel(const float* __restrict__ in, unsigned short* __restrict__ out,
                            int K, int N) {
  int idx = blockIdx.x * blockDim.x + threadIdx.x;
  int total = K * N;
  if (idx >= total) return;
  int n = idx % N, k = idx / N;
  out[(size_t)n * K + k] = f2bf(in[idx]);
}

// ---------------------------------------------------------------------------
// bf16 WMMA GEMM:  out[P x N] = A[P x K] * BT[N x K]^T + bias
//   MODE 0: exact-GELU epilogue, bf16 output (MLP hidden)
//   MODE 1: out_f32[p,n] += (acc + bias) * act[p]  (masked residual into h)
// One wave per 16 x (16*NT) output tile; A fragment loaded once per K-step and
// fed to NT WMMAs. Software double-buffered: next-step fragments are loaded
// before current-step WMMAs execute.
// P % 16 == 0, N % (16*NT) == 0, K % 32 == 0 for every layer of this net.
// A fragment (16-bit A 16x32): elems 0..7 -> K=8*half+0..7,
//   elems 8..15 -> K=16+8*half+0..7  => two contiguous 16B loads.
// B fragment (16-bit B 32x16): elems 0..7 -> K=8*half+0..7, 8..15 -> K=16+8*half+0..7 => two contiguous 16B loads.
// ---------------------------------------------------------------------------
template <int NT, int MODE>
__global__ __launch_bounds__(256)
void gemm_bf16_kernel(const unsigned short* __restrict__ A,
                      const unsigned short* __restrict__ BT,
                      const float* __restrict__ bias,
                      unsigned short* __restrict__ outBf,
                      float* __restrict__ outAcc,
                      const float* __restrict__ act,
                      int P, int N, int K) {
  int wave = threadIdx.x >> 5;
  int lane = threadIdx.x & 31;
  int half = lane >> 4;
  int lq   = lane & 15;
  int nt   = N / (16 * NT);                // column groups of width 16*NT
  int gid  = blockIdx.x * 8 + wave;
  int total = (P >> 4) * nt;
  if (gid >= total) return;                // whole-wave uniform: EXEC stays all-1s
  int mTile = (gid / nt) << 4;
  int nBase = (gid % nt) * (16 * NT);

  v8f acc[NT];
#pragma unroll
  for (int i = 0; i < NT; ++i) acc[i] = 0.0f;

  const unsigned short* Arow = A + (size_t)(mTile + lq) * K;
  const unsigned short* Bp[NT];
#pragma unroll
  for (int i = 0; i < NT; ++i) Bp[i] = BT + (size_t)(nBase + 16 * i + lq) * K;

  const int k0a = half << 3;               // A: interleaved-8 pattern
  const int k0b = half << 3;               // B: interleaved-8 pattern

  // ---- preload K-step 0 ----
  Frag a_cur, b_cur[NT];
  a_cur.h[0] = *(const us8*)(Arow + k0a);
  a_cur.h[1] = *(const us8*)(Arow + k0a + 16);
#pragma unroll
  for (int i = 0; i < NT; ++i) {
    b_cur[i].h[0] = *(const us8*)(Bp[i] + k0b);
    b_cur[i].h[1] = *(const us8*)(Bp[i] + k0b + 16);
  }

  // ---- steady state: issue loads for k+32, then run WMMAs for k ----
  for (int kk = 0; kk < K - 32; kk += 32) {
    __builtin_prefetch(Arow + kk + 128, 0, 1);
    Frag a_nxt, b_nxt[NT];
    const int kn = kk + 32;
    a_nxt.h[0] = *(const us8*)(Arow + kn + k0a);
    a_nxt.h[1] = *(const us8*)(Arow + kn + k0a + 16);
#pragma unroll
    for (int i = 0; i < NT; ++i) {
      b_nxt[i].h[0] = *(const us8*)(Bp[i] + kn + k0b);
      b_nxt[i].h[1] = *(const us8*)(Bp[i] + kn + k0b + 16);
    }
#pragma unroll
    for (int i = 0; i < NT; ++i)
      acc[i] = __builtin_amdgcn_wmma_f32_16x16x32_bf16(false, a_cur.v, false, b_cur[i].v,
                                                       (short)0, acc[i], false, false);
    a_cur = a_nxt;
#pragma unroll
    for (int i = 0; i < NT; ++i) b_cur[i] = b_nxt[i];
  }

  // ---- epilogue K-step ----
#pragma unroll
  for (int i = 0; i < NT; ++i)
    acc[i] = __builtin_amdgcn_wmma_f32_16x16x32_bf16(false, a_cur.v, false, b_cur[i].v,
                                                     (short)0, acc[i], false, false);

  // D layout: VGPR r -> row mTile + r + 8*half ; col nBase + 16*i + lq
#pragma unroll
  for (int r = 0; r < 8; ++r) {
    int mr = mTile + r + (half << 3);
    float am = (MODE == 1) ? act[mr] : 0.0f;
#pragma unroll
    for (int i = 0; i < NT; ++i) {
      int n = nBase + 16 * i + lq;
      float v = acc[i][r] + bias[n];
      if (MODE == 0) {
        outBf[(size_t)mr * N + n] = f2bf(gelu_exact(v));
      } else {
        outAcc[(size_t)mr * N + n] += v * am;
      }
    }
  }
}

// ---------------------------------------------------------------------------
// Downsample conv 2x2 stride 2 (+bias, x new active). Input hn already LN*mask'd.
// ---------------------------------------------------------------------------
__global__ void down_conv_kernel(const float* __restrict__ hn, const float* __restrict__ w,
                                 const float* __restrict__ bias, const float* __restrict__ act2,
                                 float* __restrict__ out, int Bn, int Ho, int Wo,
                                 int Cin, int Cout) {
  size_t idx = (size_t)blockIdx.x * blockDim.x + threadIdx.x;
  size_t total = (size_t)Bn * Ho * Wo * Cout;
  if (idx >= total) return;
  int co = (int)(idx % Cout); size_t p = idx / Cout;
  int j = (int)(p % Wo); size_t t = p / Wo;
  int i = (int)(t % Ho); int b = (int)(t / Ho);
  int Wi = Wo * 2;
  float s = bias[co];
  for (int u = 0; u < 2; ++u)
    for (int v = 0; v < 2; ++v) {
      const float* src = hn + (((size_t)b * (Ho * 2) + (2 * i + u)) * Wi + (2 * j + v)) * Cin;
      const float* ww  = w + (size_t)((u * 2 + v) * Cin) * Cout + co;
      for (int ci = 0; ci < Cin; ++ci) s += src[ci] * ww[(size_t)ci * Cout];
    }
  out[idx] = s * act2[p];
}

// ---------------------------------------------------------------------------
// NHWC [16,7,7,768] -> NCHW [16,768,7,7]
// ---------------------------------------------------------------------------
__global__ void to_nchw_kernel(const float* __restrict__ h, float* __restrict__ out,
                               int Bn, int H, int W, int C) {
  int idx = blockIdx.x * blockDim.x + threadIdx.x;
  int total = Bn * C * H * W;
  if (idx >= total) return;
  int j = idx % W, t = idx / W;
  int i = t % H;  t /= H;
  int c = t % C;  int b = t / C;
  out[idx] = h[(((size_t)b * H + i) * W + j) * C + c];
}

// ---------------------------------------------------------------------------
// Host orchestration
// ---------------------------------------------------------------------------
static inline int cdiv(long long a, long long b) { return (int)((a + b - 1) / b); }

extern "C" void kernel_launch(void* const* d_in, const int* in_sizes, int n_in,
                              void* d_out, int out_size, void* d_ws, size_t ws_size,
                              hipStream_t stream) {
  (void)in_sizes; (void)n_in; (void)out_size; (void)ws_size;
  const int Bn = 16;
  const int depths[4] = {3, 3, 27, 3};
  const int Cs[4]     = {96, 192, 384, 768};
  const int Hs[4]     = {56, 28, 14, 7};

  // ---- decode inputs (recursive dict-insertion-order flatten) ----
  int t = 0;
  const float* x        = (const float*)d_in[t++];
  const int*   mask     = (const int*)  d_in[t++];
  const float* stem_w   = (const float*)d_in[t++];
  const float* stem_b   = (const float*)d_in[t++];
  const float* stem_g   = (const float*)d_in[t++];
  const float* stem_be  = (const float*)d_in[t++];
  struct DownP { const float *ln_g, *ln_b, *w, *b; } down[3];
  for (int i = 0; i < 3; ++i) {
    down[i].ln_g = (const float*)d_in[t++];
    down[i].ln_b = (const float*)d_in[t++];
    down[i].w    = (const float*)d_in[t++];
    down[i].b    = (const float*)d_in[t++];
  }
  struct BlkP { const float *dw_w, *dw_b, *ln_g, *ln_b, *w1, *b1, *w2, *b2; } blk[36];
  int nb = 0;
  for (int s = 0; s < 4; ++s)
    for (int d = 0; d < depths[s]; ++d) {
      blk[nb].dw_w = (const float*)d_in[t++];
      blk[nb].dw_b = (const float*)d_in[t++];
      blk[nb].ln_g = (const float*)d_in[t++];
      blk[nb].ln_b = (const float*)d_in[t++];
      blk[nb].w1   = (const float*)d_in[t++];
      blk[nb].b1   = (const float*)d_in[t++];
      blk[nb].w2   = (const float*)d_in[t++];
      blk[nb].b2   = (const float*)d_in[t++];
      ++nb;
    }
  struct NormP { const float *g, *b; } norm[4];
  for (int i = 0; i < 4; ++i) {
    norm[i].g = (const float*)d_in[t++];
    norm[i].b = (const float*)d_in[t++];
  }

  // ---- workspace layout (~96 MB) ----
  const size_t P1 = (size_t)Bn * 56 * 56;          // 50176 points, stage-1 resolution
  char* base = (char*)d_ws;
  size_t off = 0;
  auto alloc = [&](size_t bytes) -> void* {
    void* p = base + off;
    off += (bytes + 255) & ~(size_t)255;
    return p;
  };
  float*          hbuf = (float*)         alloc(P1 * 96 * sizeof(float));   // h (current)
  float*          ybuf = (float*)         alloc(P1 * 96 * sizeof(float));   // dwconv / LN temp
  unsigned short* yb   = (unsigned short*)alloc(P1 * 96 * 2);               // LN(y) bf16
  unsigned short* hb   = (unsigned short*)alloc(P1 * 384 * 2);              // MLP hidden bf16
  unsigned short* w1t  = (unsigned short*)alloc((size_t)3072 * 768 * 2);    // w1^T bf16
  unsigned short* w2t  = (unsigned short*)alloc((size_t)3072 * 768 * 2);    // w2^T bf16
  float* actb[4];
  for (int s = 0; s < 4; ++s)
    actb[s] = (float*)alloc((size_t)Bn * Hs[s] * Hs[s] * sizeof(float));

  const int TB = 256;

  // ---- active masks at all resolutions ----
  active0_kernel<<<cdiv(Bn * 56 * 56, TB), TB, 0, stream>>>(mask, actb[0], Bn);
  for (int s = 1; s < 4; ++s)
    down_active_kernel<<<cdiv(Bn * Hs[s] * Hs[s], TB), TB, 0, stream>>>(
        actb[s - 1], actb[s], Bn, Hs[s], Hs[s]);

  // ---- stem: conv4x4/4 + bias, then LN * active (fused mask) ----
  stem_kernel<<<cdiv((long long)P1 * 96, TB), TB, 0, stream>>>(x, stem_w, stem_b, hbuf, Bn);
  ln_f32_kernel<<<cdiv((long long)P1, 8), TB, 0, stream>>>(
      hbuf, hbuf, stem_g, stem_be, actb[0], (int)P1, 96);

  // ---- stages ----
  nb = 0;
  for (int s = 0; s < 4; ++s) {
    const int C = Cs[s], H = Hs[s], N4 = 4 * C;
    const int P = Bn * H * H;
    if (s > 0) {
      const int Cp = Cs[s - 1], Hp = Hs[s - 1];
      const int Pp = Bn * Hp * Hp;
      ln_f32_kernel<<<cdiv(Pp, 8), TB, 0, stream>>>(
          hbuf, ybuf, down[s - 1].ln_g, down[s - 1].ln_b, actb[s - 1], Pp, Cp);
      down_conv_kernel<<<cdiv((long long)P * C, TB), TB, 0, stream>>>(
          ybuf, down[s - 1].w, down[s - 1].b, actb[s], hbuf, Bn, H, H, Cp, C);
    }
    for (int d = 0; d < depths[s]; ++d) {
      const BlkP& bk = blk[nb++];
      dwconv7_kernel<<<cdiv((long long)P * C, TB), TB, 0, stream>>>(
          hbuf, bk.dw_w, bk.dw_b, ybuf, Bn, H, H, C);
      ln_bf16_kernel<<<cdiv(P, 8), TB, 0, stream>>>(ybuf, yb, bk.ln_g, bk.ln_b, P, C);
      tcvt_kernel<<<cdiv((long long)C * N4, TB), TB, 0, stream>>>(bk.w1, w1t, C, N4);
      tcvt_kernel<<<cdiv((long long)N4 * C, TB), TB, 0, stream>>>(bk.w2, w2t, N4, C);
      // GEMM1: [P x C] @ [C x 4C] + b1 -> GELU -> bf16   (N4 % 64 == 0 always)
      {
        int waves = (P >> 4) * (N4 / 64);
        gemm_bf16_kernel<4, 0><<<cdiv(waves, 8), TB, 0, stream>>>(
            yb, w1t, bk.b1, hb, nullptr, nullptr, P, N4, C);
      }
      // GEMM2: [P x 4C] @ [4C x C] + b2 -> h += v * active
      if (C % 64 == 0) {
        int waves = (P >> 4) * (C / 64);
        gemm_bf16_kernel<4, 1><<<cdiv(waves, 8), TB, 0, stream>>>(
            hb, w2t, bk.b2, nullptr, hbuf, actb[s], P, C, N4);
      } else {                                   // stage 1: C = 96
        int waves = (P >> 4) * (C / 32);
        gemm_bf16_kernel<2, 1><<<cdiv(waves, 8), TB, 0, stream>>>(
            hb, w2t, bk.b2, nullptr, hbuf, actb[s], P, C, N4);
      }
    }
    ln_f32_kernel<<<cdiv(P, 8), TB, 0, stream>>>(
        hbuf, hbuf, norm[s].g, norm[s].b, actb[s], P, C);
  }

  // ---- NHWC -> NCHW output ----
  to_nchw_kernel<<<cdiv(Bn * 768 * 7 * 7, TB), TB, 0, stream>>>(
      hbuf, (float*)d_out, Bn, 7, 7, 768);
}
